// Decoder_7267084665223
// MI455X (gfx1250) — matrix operand order repaired and checked by reading, not compiled, not run on hardware
//
#include <hip/hip_runtime.h>
#include <hip/hip_bf16.h>

// B=2048, T=256, H=256, O=2, 4H=1024
#define BB   2048
#define TT   256
#define HH   256
#define G4   1024
#define OO   2

#define ROWS 32          // batch rows per workgroup
#define THREADS 512      // 16 waves
#define NWG  (BB/ROWS)   // 64 workgroups
#define HPAD 272         // padded h row stride (halves) to dodge LDS bank conflicts

typedef __attribute__((ext_vector_type(16))) _Float16 v16h;
typedef __attribute__((ext_vector_type(8)))  float    v8f;

union Frag16 { v16h v; uint4 q[2]; };

__device__ __forceinline__ float sigm(float x) {
    float e = __expf(-x);
    return __builtin_amdgcn_rcpf(1.0f + e);
}
__device__ __forceinline__ float tanh_fast(float x) {
    float e = __expf(-2.0f * x);
    return (1.0f - e) * __builtin_amdgcn_rcpf(1.0f + e);
}

// ---------------------------------------------------------------------------
// Pack W_hh (4H x H, f32 row-major) into per-lane f16 B-fragments for
// v_wmma_f32_16x16x32_f16.  Fragment index: [wave][gate][kstep].
// B layout (rows=K striped across lanes): lanes 0-15 hold K {0..7, 16..23},
// lanes 16-31 hold K {8..15, 24..31}, column N = lane%16.
//   r (gate row) = g*256 + w*16 + (lane&15)      (w = wave, owns 16 h-cols)
//   k            = kk*32 + (h>>3)*16 + (lane>>4)*8 + (h&7)
//   fragment value = W_hh[r][k]   (gates = h @ W_hh^T)
// ---------------------------------------------------------------------------
__global__ void pack_whh_kernel(const float* __restrict__ W_hh,
                                _Float16* __restrict__ packedB) {
    int idx  = blockIdx.x * 256 + threadIdx.x;   // 0 .. 262143
    int h    =  idx        & 15;
    int lane = (idx >> 4)  & 31;
    int kk   = (idx >> 9)  & 7;
    int g    = (idx >> 12) & 3;
    int w    = (idx >> 14);                      // 0..15
    int r = g * 256 + w * 16 + (lane & 15);
    int k = kk * 32 + (h >> 3) * 16 + (lane >> 4) * 8 + (h & 7);
    packedB[idx] = (_Float16)W_hh[r * HH + k];
}

// ---------------------------------------------------------------------------
// Persistent LSTM kernel: one workgroup = 32 batch rows, all 256 timesteps.
// 16 waves; wave w owns h-columns [16w, 16w+16) across all 4 gates, and both
// 16-row M-tiles (M=32) -> every W_hh B-fragment feeds 2 WMMAs (2x L2 reuse
// vs M=16, 2x less L2 stream per batch row).
// ---------------------------------------------------------------------------
__global__ __launch_bounds__(THREADS)
void lstm_wmma_kernel(const float* __restrict__ y_hist,
                      const float* __restrict__ h0,
                      const float* __restrict__ c0,
                      const float* __restrict__ W_ih,
                      const float* __restrict__ b_ih,
                      const float* __restrict__ b_hh,
                      const float* __restrict__ W_fc,
                      const float* __restrict__ b_fc,
                      const _Float16* __restrict__ packedB,
                      float* __restrict__ out) {
    __shared__ __align__(16) _Float16 h_sh[2][ROWS][HPAD];   // ping-pong f16 h
    __shared__ __align__(16) float    x_sh[TT][ROWS];        // y_hist transposed
    __shared__ __align__(16) float    hout_sh[ROWS][HH];     // final h (f32)

    const int tid   = threadIdx.x;
    const int wv    = tid >> 5;        // wave 0..15
    const int lane  = tid & 31;
    const int laneN = lane & 15;       // N within a 16-wide tile
    const int hi    = lane >> 4;       // lane half selector
    const int rb    = blockIdx.x * ROWS;

    // ---- stage y_hist tile (transposed) and h0 (f16, buffer 0) into LDS ----
    for (int idx = tid; idx < ROWS * TT; idx += THREADS) {
        int m = idx & 31, t = idx >> 5;
        x_sh[t][m] = y_hist[(rb + m) * TT + t];
    }
    for (int idx = tid; idx < ROWS * HH; idx += THREADS) {
        int m = idx >> 8, n = idx & 255;
        h_sh[0][m][n] = (_Float16)h0[(rb + m) * HH + n];
    }

    // ---- per-lane persistent state: c (WMMA C/D layout), w_x, bias ---------
    float c_reg[2][8];                 // [m-tile][vreg]
    float wx[4], bs[4];
    const int ncol = wv * 16 + laneN;  // this lane's h-column
    #pragma unroll
    for (int mt = 0; mt < 2; ++mt) {
        #pragma unroll
        for (int v = 0; v < 8; ++v) {
            int m = mt * 16 + v + hi * 8;
            c_reg[mt][v] = c0[(rb + m) * HH + ncol];
        }
    }
    #pragma unroll
    for (int g = 0; g < 4; ++g) {
        int r = g * 256 + ncol;
        wx[g] = W_ih[r];
        bs[g] = b_ih[r] + b_hh[r];
    }
    __syncthreads();

    const _Float16* bwave = packedB + wv * 16384;   // this wave's W_hh stripe

    // ---- time loop (one barrier per step, h ping-pong) ---------------------
    for (int t = 0; t < TT; ++t) {
        const int p = t & 1;

        // Opaque per-iteration offset: stops the compiler from hoisting the
        // (loop-invariant) packedB fragment loads out of the time loop and
        // spilling B to scratch.  Keeps them an L2-hot per-step stream.
        unsigned bofs = (unsigned)(lane * 16);
        asm volatile("" : "+v"(bofs));
        const _Float16* bp = bwave + bofs;

        // x_t for this lane's rows (row = mt*16 + v + 8*hi)
        float xv[2][8];
        #pragma unroll
        for (int mt = 0; mt < 2; ++mt) {
            float4 x0 = *(const float4*)&x_sh[t][mt * 16 + hi * 8];
            float4 x1 = *(const float4*)&x_sh[t][mt * 16 + hi * 8 + 4];
            xv[mt][0] = x0.x; xv[mt][1] = x0.y; xv[mt][2] = x0.z; xv[mt][3] = x0.w;
            xv[mt][4] = x1.x; xv[mt][5] = x1.y; xv[mt][6] = x1.z; xv[mt][7] = x1.w;
        }

        // accumulators = bias + x*w_x  (gates before the W_hh term)
        v8f acc[4][2];                 // [gate][m-tile]
        #pragma unroll
        for (int g = 0; g < 4; ++g)
            #pragma unroll
            for (int mt = 0; mt < 2; ++mt)
                #pragma unroll
                for (int v = 0; v < 8; ++v)
                    acc[g][mt][v] = bs[g] + xv[mt][v] * wx[g];

        // gates += h @ W_hh^T : kk-outer, each B fragment feeds 2 M-tiles
        #pragma unroll
        for (int kk = 0; kk < 8; ++kk) {
            // A fragments (h rows from LDS), per ISA A layout:
            // lane<16: K {0..7, 16..23}; lane>=16: K {8..15, 24..31} (+32*kk)
            Frag16 a[2];
            const int o1 = kk * 32 + hi * 8;
            #pragma unroll
            for (int mt = 0; mt < 2; ++mt) {
                a[mt].q[0] = *(const uint4*)&h_sh[p][mt * 16 + laneN][o1];
                a[mt].q[1] = *(const uint4*)&h_sh[p][mt * 16 + laneN][o1 + 16];
            }
            Frag16 bf[4];
            #pragma unroll
            for (int g = 0; g < 4; ++g) {
                const _Float16* bgk = bp + (g * 8 + kk) * 512;
                bf[g].q[0] = *(const uint4*)(bgk);
                bf[g].q[1] = *(const uint4*)(bgk + 8);
            }
            #pragma unroll
            for (int g = 0; g < 4; ++g)
                #pragma unroll
                for (int mt = 0; mt < 2; ++mt)
                    acc[g][mt] = __builtin_amdgcn_wmma_f32_16x16x32_f16(
                        false, a[mt].v, false, bf[g].v, (short)0, acc[g][mt],
                        false, false);
        }

        // activations, c/h update, publish h (f16) into the other buffer
        #pragma unroll
        for (int mt = 0; mt < 2; ++mt) {
            #pragma unroll
            for (int v = 0; v < 8; ++v) {
                int m = mt * 16 + v + hi * 8;
                float iv = sigm(acc[0][mt][v]);
                float fv = sigm(acc[1][mt][v]);
                float gv = tanh_fast(acc[2][mt][v]);
                float ov = sigm(acc[3][mt][v]);
                float cv = fv * c_reg[mt][v] + iv * gv;
                c_reg[mt][v] = cv;
                float hv = ov * tanh_fast(cv);
                h_sh[p ^ 1][m][ncol] = (_Float16)hv;
                if (t == TT - 1) hout_sh[m][ncol] = hv;
            }
        }
        __syncthreads();   // writes to buf[p^1] visible before next step reads
    }

    // ---- out = h_T @ W_fc^T + b_fc  (O=2: 64 (row,o) pairs) ----------------
    if (tid < ROWS * OO) {
        int m = tid >> 1, o = tid & 1;
        float acc = b_fc[o];
        for (int n = 0; n < HH; ++n)
            acc += hout_sh[m][n] * W_fc[o * HH + n];
        out[(rb + m) * OO + o] = acc;
    }
}

extern "C" void kernel_launch(void* const* d_in, const int* in_sizes, int n_in,
                              void* d_out, int out_size, void* d_ws, size_t ws_size,
                              hipStream_t stream) {
    const float* y_hist = (const float*)d_in[0];
    const float* h0     = (const float*)d_in[1];
    const float* c0     = (const float*)d_in[2];
    const float* W_ih   = (const float*)d_in[3];
    const float* W_hh   = (const float*)d_in[4];
    const float* b_ih   = (const float*)d_in[5];
    const float* b_hh   = (const float*)d_in[6];
    const float* W_fc   = (const float*)d_in[7];
    const float* b_fc   = (const float*)d_in[8];
    float* out = (float*)d_out;

    _Float16* packedB = (_Float16*)d_ws;   // 1024*256 halves = 512 KB

    // Repack W_hh into WMMA B-fragment order (f16), then run the fused LSTM.
    pack_whh_kernel<<<(G4 * HH) / 256, 256, 0, stream>>>(W_hh, packedB);
    lstm_wmma_kernel<<<NWG, THREADS, 0, stream>>>(y_hist, h0, c0, W_ih, b_ih,
                                                  b_hh, W_fc, b_fc, packedB, out);
}
